// CARRVProj_59261958750431
// MI455X (gfx1250) — matrix operand order repaired and checked by reading, not compiled, not run on hardware
//
#include <hip/hip_runtime.h>
#include <hip/hip_bf16.h>

typedef __attribute__((ext_vector_type(16))) __bf16 v16bf;
typedef __attribute__((ext_vector_type(8)))  __bf16 v8bf;
typedef __attribute__((ext_vector_type(8)))  float  v8f;
typedef __hip_bfloat16 bf16_t;

#define BM 128
#define BN 128
#define BK 32
#define LDSS 40   // padded LDS row stride (elements): 80B, keeps 16B chunk alignment

#if __has_builtin(__builtin_amdgcn_global_load_async_to_lds_b128) && \
    __has_builtin(__builtin_amdgcn_s_wait_asynccnt)
#define USE_ASYNC_LDS 1
#else
#define USE_ASYNC_LDS 0
#endif

typedef int v4i_vs __attribute__((vector_size(16)));
typedef __attribute__((address_space(1))) v4i_vs* as1_v4i_ptr;
typedef __attribute__((address_space(3))) v4i_vs* as3_v4i_ptr;

__device__ __forceinline__ void async_copy16(const bf16_t* g, bf16_t* l) {
#if USE_ASYNC_LDS
    __builtin_amdgcn_global_load_async_to_lds_b128(
        (as1_v4i_ptr)(void*)g,
        (as3_v4i_ptr)(void*)l, 0, 0);
#endif
}

// ---------------- cast / pack kernels ----------------

__global__ void k_cast(const float* __restrict__ src, bf16_t* __restrict__ dst, int n) {
    int i = blockIdx.x * blockDim.x + threadIdx.x;
    int stride = gridDim.x * blockDim.x;
    for (; i < n; i += stride) dst[i] = __float2bfloat16(src[i]);
}

// Wcomb: 128 rows x 1024. rows 0..7 = Wg, rows 8..71 = Wp (flattened (e,p)), rest zero.
__global__ void k_build_wcomb(const float* __restrict__ Wg, const float* __restrict__ Wp,
                              bf16_t* __restrict__ dst) {
    int i = blockIdx.x * blockDim.x + threadIdx.x;   // 128*1024 total
    if (i >= 128 * 1024) return;
    int row = i >> 10;
    int d   = i & 1023;
    float v = 0.0f;
    if (row < 8)       v = Wg[row * 1024 + d];
    else if (row < 72) v = Wp[(row - 8) * 1024 + d];
    dst[i] = __float2bfloat16(v);
}

// W2all: (1024 x 256), W2all[d][e*32+i] = W2[e][d][i]   (W2 is (E, DV, INNER))
__global__ void k_build_w2all(const float* __restrict__ W2, bf16_t* __restrict__ dst) {
    int i = blockIdx.x * blockDim.x + threadIdx.x;   // 1024*256 total
    if (i >= 1024 * 256) return;
    int d  = i >> 8;
    int j  = i & 255;
    int e  = j >> 5;
    int ii = j & 31;
    dst[i] = __float2bfloat16(W2[(e * 1024 + d) * 32 + ii]);
}

// ---------------- routing kernel ----------------

__global__ void k_route(const float* __restrict__ SC, const float* __restrict__ bg,
                        const float* __restrict__ alpha,
                        const float* __restrict__ g_r, const float* __restrict__ b_r,
                        const float* __restrict__ g_c, const float* __restrict__ b_c,
                        float* __restrict__ we, int n) {
    int t = blockIdx.x * blockDim.x + threadIdx.x;
    if (t >= n) return;
    const float* sc = SC + (size_t)t * 128;
    float r[8], c[8], s[8];
    #pragma unroll
    for (int e = 0; e < 8; ++e) r[e] = sc[e] + bg[e];
    #pragma unroll
    for (int e = 0; e < 8; ++e) {
        float acc = 0.f;
        #pragma unroll
        for (int p = 0; p < 8; ++p) { float v = sc[8 + e * 8 + p]; acc += v * v; }
        c[e] = sqrtf(acc * 0.125f);   // sqrt(sum)/sqrt(P), P=8
    }
    float mr = 0.f, mc = 0.f;
    #pragma unroll
    for (int e = 0; e < 8; ++e) { mr += r[e]; mc += c[e]; }
    mr *= 0.125f; mc *= 0.125f;
    float vr = 0.f, vc = 0.f;
    #pragma unroll
    for (int e = 0; e < 8; ++e) {
        float dr = r[e] - mr, dc = c[e] - mc;
        vr += dr * dr; vc += dc * dc;
    }
    vr *= 0.125f; vc *= 0.125f;
    float ir = rsqrtf(vr + 1e-5f), ic = rsqrtf(vc + 1e-5f);
    float sa = 1.0f / (1.0f + expf(-alpha[0]));
    #pragma unroll
    for (int e = 0; e < 8; ++e)
        s[e] = (r[e] - mr) * ir * g_r[e] + b_r[e] + sa * ((c[e] - mc) * ic * g_c[e] + b_c[e]);
    // top-2, jax tie-breaking: first strict max
    int i0 = 0;
    #pragma unroll
    for (int e = 1; e < 8; ++e) if (s[e] > s[i0]) i0 = e;
    int i1 = (i0 == 0) ? 1 : 0;
    #pragma unroll
    for (int e = 0; e < 8; ++e) if (e != i0 && s[e] > s[i1]) i1 = e;
    float s0 = s[i0], s1 = s[i1];
    float e1 = expf(s1 - s0);                 // s0 >= s1
    float w0 = 1.0f / (1.0f + e1);
    float w1 = e1 * w0;
    float* w = we + (size_t)t * 8;
    #pragma unroll
    for (int e = 0; e < 8; ++e) w[e] = 0.f;
    w[i0] = w0; w[i1] = w1;
}

// ---------------- generic bf16 WMMA GEMM:  C(M x N) = A(M x K) * B(N x K)^T ----------------
// Block tile 128x128, 8 waves, wave tile 32x64 (2x4 WMMA frags), double-buffered LDS,
// one barrier per K-step. Async global->LDS (ASYNCcnt) when the builtin exists.
// mode 0: C32 = acc + bias[col]; also CBF = bf16(C32)            (V projection)
// mode 1: C32 = acc                                              (router scores)
// mode 2: CBF = bf16( we[row][col/32] * silu(acc + bias[col]) )  (expert hidden * gate)
// mode 3: C32 += acc + sum_e we[row][e]*bias[e*N+col]            (combine, bias = b2)

__global__ __launch_bounds__(256) void k_gemm(
    const bf16_t* __restrict__ A, const bf16_t* __restrict__ B,
    int K, int N, int mode,
    float* __restrict__ C32, bf16_t* __restrict__ CBF,
    const float* __restrict__ bias, const float* __restrict__ we) {

    __shared__ bf16_t sA[2][BM * LDSS];
    __shared__ bf16_t sB[2][BN * LDSS];

    const int tid  = threadIdx.x;
    const int m0   = blockIdx.x * BM;
    const int n0   = blockIdx.y * BN;
    const int wave = tid >> 5;
    const int lane = tid & 31;
    const int wm = wave >> 1;          // 0..3  (32-row strip)
    const int wn = wave & 1;           // 0..1  (64-col strip)
    const int lr = lane & 15;
    const int lh = lane >> 4;

    v8f acc[2][4] = {};

    const bf16_t* Aptr = A + (size_t)m0 * K;
    const bf16_t* Bptr = B + (size_t)n0 * K;

    // per-thread staging chunks: A tile 128x32 -> 512 x 16B, B tile same; 2+2 per thread
    const int srow0 = tid >> 2;              // chunk row for c=0 (0..63)
    const int sk8   = (tid & 3) << 3;        // 0,8,16,24
    const int nk = K / BK;

#if USE_ASYNC_LDS
    // prologue: tile 0 -> buffer 0
    #pragma unroll
    for (int cc = 0; cc < 2; ++cc) {
        int row = srow0 + cc * 64;
        async_copy16(&Aptr[(size_t)row * K + sk8], &sA[0][row * LDSS + sk8]);
        async_copy16(&Bptr[(size_t)row * K + sk8], &sB[0][row * LDSS + sk8]);
    }
#else
    v8bf ra[2], rb[2];
    #pragma unroll
    for (int cc = 0; cc < 2; ++cc) {
        int row = srow0 + cc * 64;
        ra[cc] = *(const v8bf*)&Aptr[(size_t)row * K + sk8];
        rb[cc] = *(const v8bf*)&Bptr[(size_t)row * K + sk8];
    }
#endif

    for (int k = 0; k < nk; ++k) {
        const int cur = k & 1;
        const int nxt = cur ^ 1;
#if USE_ASYNC_LDS
        __builtin_amdgcn_s_wait_asynccnt(0);   // tile k resident in sA/sB[cur]
        __syncthreads();                       // ...for every wave; also fences buf[nxt] readers
        if (k + 1 < nk) {
            int kt = (k + 1) * BK;
            #pragma unroll
            for (int cc = 0; cc < 2; ++cc) {
                int row = srow0 + cc * 64;
                async_copy16(&Aptr[(size_t)row * K + kt + sk8], &sA[nxt][row * LDSS + sk8]);
                async_copy16(&Bptr[(size_t)row * K + kt + sk8], &sB[nxt][row * LDSS + sk8]);
            }
        }
#else
        #pragma unroll
        for (int cc = 0; cc < 2; ++cc) {
            int row = srow0 + cc * 64;
            *(v8bf*)&sA[cur][row * LDSS + sk8] = ra[cc];
            *(v8bf*)&sB[cur][row * LDSS + sk8] = rb[cc];
        }
        if (k + 1 < nk) {
            int kt = (k + 1) * BK;
            #pragma unroll
            for (int cc = 0; cc < 2; ++cc) {
                int row = srow0 + cc * 64;
                ra[cc] = *(const v8bf*)&Aptr[(size_t)row * K + kt + sk8];
                rb[cc] = *(const v8bf*)&Bptr[(size_t)row * K + kt + sk8];
            }
        }
        __syncthreads();
#endif
        // fragments
        v16bf afrag[2], bfrag[4];
        #pragma unroll
        for (int mi = 0; mi < 2; ++mi) {
            const bf16_t* base = &sA[cur][(wm * 32 + mi * 16 + lr) * LDSS];
            v8bf lo = *(const v8bf*)&base[lh * 8];
            v8bf hi = *(const v8bf*)&base[16 + lh * 8];
            #pragma unroll
            for (int i = 0; i < 8; ++i) { afrag[mi][i] = lo[i]; afrag[mi][8 + i] = hi[i]; }
        }
        #pragma unroll
        for (int ni = 0; ni < 4; ++ni) {
            const bf16_t* base = &sB[cur][(wn * 64 + ni * 16 + lr) * LDSS];
            v8bf lo = *(const v8bf*)&base[lh * 8];
            v8bf hi = *(const v8bf*)&base[16 + lh * 8];
            #pragma unroll
            for (int i = 0; i < 8; ++i) { bfrag[ni][i] = lo[i]; bfrag[ni][8 + i] = hi[i]; }
        }
        #pragma unroll
        for (int mi = 0; mi < 2; ++mi)
            #pragma unroll
            for (int ni = 0; ni < 4; ++ni)
                acc[mi][ni] = __builtin_amdgcn_wmma_f32_16x16x32_bf16(
                    false, afrag[mi], false, bfrag[ni], (short)0, acc[mi][ni], false, false);
#if !USE_ASYNC_LDS
        __syncthreads();
#endif
    }

    // epilogue: element (vgpr r, lane) -> M = r + 8*lh, N = lr within 16x16 frag
    #pragma unroll
    for (int mi = 0; mi < 2; ++mi)
        #pragma unroll
        for (int ni = 0; ni < 4; ++ni)
            #pragma unroll
            for (int r = 0; r < 8; ++r) {
                int row = m0 + wm * 32 + mi * 16 + r + lh * 8;
                int col = n0 + wn * 64 + ni * 16 + lr;
                float v = acc[mi][ni][r];
                size_t idx = (size_t)row * N + col;
                if (mode == 0) {
                    v += bias[col];
                    C32[idx] = v;
                    CBF[idx] = __float2bfloat16(v);
                } else if (mode == 1) {
                    C32[idx] = v;
                } else if (mode == 2) {
                    v += bias[col];
                    float sil = v / (1.0f + expf(-v));
                    float w = we[(size_t)row * 8 + (col >> 5)];
                    CBF[idx] = __float2bfloat16(sil * w);
                } else {
                    float b = 0.f;
                    #pragma unroll
                    for (int e = 0; e < 8; ++e)
                        b += we[(size_t)row * 8 + e] * bias[e * N + col];
                    C32[idx] += v + b;
                }
            }
}

// ---------------- launcher ----------------

extern "C" void kernel_launch(void* const* d_in, const int* in_sizes, int n_in,
                              void* d_out, int out_size, void* d_ws, size_t ws_size,
                              hipStream_t stream) {
    const float* x     = (const float*)d_in[0];
    const float* Wv    = (const float*)d_in[1];
    const float* bv    = (const float*)d_in[2];
    const float* Wg    = (const float*)d_in[3];
    const float* bg    = (const float*)d_in[4];
    const float* Wp    = (const float*)d_in[5];
    const float* alpha = (const float*)d_in[6];
    const float* g_r   = (const float*)d_in[7];
    const float* b_r   = (const float*)d_in[8];
    const float* g_c   = (const float*)d_in[9];
    const float* b_c   = (const float*)d_in[10];
    const float* W1    = (const float*)d_in[11];
    const float* b1    = (const float*)d_in[12];
    const float* W2    = (const float*)d_in[13];
    const float* b2    = (const float*)d_in[14];
    float* out = (float*)d_out;

    constexpr int NT = 4 * 4096;     // tokens
    constexpr int DV = 1024;

    char* p = (char*)d_ws;
    auto take = [&](size_t bytes) { char* q = p; p += (bytes + 255) & ~(size_t)255; return q; };
    bf16_t* xb    = (bf16_t*)take((size_t)NT * 1024 * 2);
    bf16_t* Wvb   = (bf16_t*)take((size_t)DV * 1024 * 2);
    bf16_t* Wcomb = (bf16_t*)take((size_t)128 * 1024 * 2);
    bf16_t* W1a   = (bf16_t*)take((size_t)256 * 1024 * 2);
    bf16_t* W2a   = (bf16_t*)take((size_t)1024 * 256 * 2);
    bf16_t* Vbf   = (bf16_t*)take((size_t)NT * DV * 2);
    float*  SC    = (float*)take((size_t)NT * 128 * 4);
    float*  we    = (float*)take((size_t)NT * 8 * 4);
    bf16_t* gbuf  = (bf16_t*)take((size_t)NT * 256 * 2);

    // casts / packs
    k_cast<<<8192, 256, 0, stream>>>(x, xb, NT * 1024);
    k_cast<<<1024, 256, 0, stream>>>(Wv, Wvb, DV * 1024);
    k_cast<<<256, 256, 0, stream>>>(W1, W1a, 256 * 1024);
    k_build_wcomb<<<(128 * 1024) / 256, 256, 0, stream>>>(Wg, Wp, Wcomb);
    k_build_w2all<<<(1024 * 256) / 256, 256, 0, stream>>>(W2, W2a);

    // GEMM1: V = x @ Wv^T + bv   -> out (fp32), Vbf (bf16)
    {
        dim3 g(NT / BM, DV / BN);
        k_gemm<<<g, 256, 0, stream>>>(xb, Wvb, 1024, DV, 0, out, Vbf, bv, nullptr);
    }
    // GEMM2: SC = V @ Wcomb^T    (N = 128 padded: 8 gate + 64 probe rows)
    {
        dim3 g(NT / BM, 128 / BN);
        k_gemm<<<g, 256, 0, stream>>>(Vbf, Wcomb, 1024, 128, 1, SC, nullptr, nullptr, nullptr);
    }
    // routing -> dense we[NT][8]
    k_route<<<NT / 256, 256, 0, stream>>>(SC, bg, alpha, g_r, b_r, g_c, b_c, we, NT);

    // GEMM3: g = we * silu(V @ W1a^T + b1)   (N = 256 = E*INNER)
    {
        dim3 g(NT / BM, 256 / BN);
        k_gemm<<<g, 256, 0, stream>>>(Vbf, W1a, 1024, 256, 2, nullptr, gbuf, b1, we);
    }
    // GEMM4: out += g @ W2a^T + sum_e we*b2   (K = 256)
    {
        dim3 g(NT / BM, DV / BN);
        k_gemm<<<g, 256, 0, stream>>>(gbuf, W2a, 256, DV, 3, out, nullptr, b2, we);
    }
}
